// AMK_Block_58128087384434
// MI455X (gfx1250) — matrix operand order repaired and checked by reading, not compiled, not run on hardware
//
#include <hip/hip_runtime.h>
#include <hip/hip_bf16.h>

typedef _Float16 half_t;
typedef __attribute__((ext_vector_type(16))) _Float16 v16h;
typedef __attribute__((ext_vector_type(8)))  _Float16 v8h;
typedef __attribute__((ext_vector_type(8)))  float    v8f;

#define BN 8192        // B*N
#define DM 512         // d_model
#define DS 64          // D_SPEC
#define NN 2048        // N
#define INNER 2048
#define KCH 64         // K-chunk (halves) staged per async round

__device__ __forceinline__ int lane_id() { return threadIdx.x & 31; }
__device__ __forceinline__ int wave_id() { return threadIdx.x >> 5; }

__device__ __forceinline__ v8f wmma16(v16h a, v16h b, v8f c) {
  return __builtin_amdgcn_wmma_f32_16x16x32_f16(false, a, false, b, (short)0, c, false, false);
}

// A operand: row-major [*, ldk] f16. lanes0-15 row=lane, K = k0+[0..7], k0+16+[0..7];
// lanes16-31 row=lane-16, K = k0+8+[0..7], k0+24+[0..7].
__device__ __forceinline__ v16h load_a(const half_t* A, int ldk, int row0, int k0, int lane) {
  int row = row0 + (lane & 15);
  int hi  = lane >> 4;
  const half_t* p = A + (size_t)row * ldk + k0 + hi * 8;
  v8h a0 = *(const v8h*)p;
  v8h a1 = *(const v8h*)(p + 16);
  v16h r;
#pragma unroll
  for (int i = 0; i < 8; ++i) { r[i] = a0[i]; r[i + 8] = a1[i]; }
  return r;
}

// B operand from weight W[Nc, K] row-major (computing A @ W^T).
__device__ __forceinline__ v16h load_b(const half_t* W, int ldk, int col0, int k0, int lane) {
  int col = col0 + (lane & 15);
  int hi  = lane >> 4;
  const half_t* p = W + (size_t)col * ldk + k0 + hi * 8;
  v8h b0 = *(const v8h*)p;
  v8h b1 = *(const v8h*)(p + 16);
  v16h r;
#pragma unroll
  for (int i = 0; i < 8; ++i) { r[i] = b0[i]; r[i + 8] = b1[i]; }
  return r;
}

// A operand from an LDS-staged chunk: [64 rows][KCH halves], row stride KCH.
__device__ __forceinline__ v16h lds_a(const half_t* buf, int rt, int kk, int lane) {
  int row = rt * 16 + (lane & 15);
  int hi  = lane >> 4;
  const half_t* p = buf + row * KCH + kk + hi * 8;
  v8h a0 = *(const v8h*)p;
  v8h a1 = *(const v8h*)(p + 16);
  v16h r;
#pragma unroll
  for (int i = 0; i < 8; ++i) { r[i] = a0[i]; r[i + 8] = a1[i]; }
  return r;
}

// Async-DMA one 64-row x KCH-half A chunk into LDS (8KB = 512 x 16B pieces,
// 2 pieces per thread @ 256 threads). Tracked by ASYNCcnt (2 per wave).
__device__ __forceinline__ void stage_chunk(half_t* dst, const half_t* A, int row0,
                                            int kbase, int ldk, int tid) {
  uint32_t base = (uint32_t)(uintptr_t)dst;
#pragma unroll
  for (int j = 0; j < 2; ++j) {
    int piece = tid + j * 256;
    int row = piece >> 3, seg = piece & 7;
    const half_t* g = A + (size_t)(row0 + row) * ldk + kbase + seg * 8;
    asm volatile("global_load_async_to_lds_b128 %0, %1, off"
                 :: "v"(base + (uint32_t)piece * 16), "v"(g) : "memory");
  }
}

// ---------------- LayerNorm 1: Qn1 f32/f16/f16-transposed + q_pool ----------------
__global__ void ln1_kernel(const float* __restrict__ Qin, const float* __restrict__ g,
                           const float* __restrict__ bta, float* __restrict__ qn1f,
                           half_t* __restrict__ qn1h, half_t* __restrict__ qn1t,
                           float* __restrict__ qpool) {
  int row = blockIdx.x;            // b*N + n
  int b = row >> 11, n = row & (NN - 1);
  const float* x = Qin + (size_t)row * DM;
  __shared__ float red[8];
  float s = 0.f;
  for (int i = threadIdx.x; i < DM; i += 256) s += x[i];
  for (int m = 16; m >= 1; m >>= 1) s += __shfl_xor(s, m, 32);
  if (lane_id() == 0) red[wave_id()] = s;
  __syncthreads();
  float mu = 0.f;
  for (int i = 0; i < 8; ++i) mu += red[i];
  mu *= (1.f / DM);
  __syncthreads();
  float v = 0.f;
  for (int i = threadIdx.x; i < DM; i += 256) { float d = x[i] - mu; v += d * d; }
  for (int m = 16; m >= 1; m >>= 1) v += __shfl_xor(v, m, 32);
  if (lane_id() == 0) red[wave_id()] = v;
  __syncthreads();
  float var = 0.f;
  for (int i = 0; i < 8; ++i) var += red[i];
  float rstd = rsqrtf(var * (1.f / DM) + 1e-5f);
  for (int i = threadIdx.x; i < DM; i += 256) {
    float y = (x[i] - mu) * rstd * g[i] + bta[i];
    qn1f[(size_t)row * DM + i] = y;
    qn1h[(size_t)row * DM + i] = (half_t)y;
    qn1t[((size_t)b * DM + i) * NN + n] = (half_t)y;
    atomicAdd(&qpool[b * DM + i], y);
  }
}

// ---------------- Omega: OmT[{Q,K}][b][e][dcol] = (q_pool/N) . W_row -------------
__global__ void omega_kernel(const float* __restrict__ Wq, const float* __restrict__ Wk,
                             const float* __restrict__ qpool, half_t* __restrict__ omtQ,
                             half_t* __restrict__ omtK) {
  int idx = blockIdx.x * 256 + threadIdx.x;   // 2 * 4 * 32768
  int w   = idx >> 17;
  int rem = idx & 131071;
  int b   = rem >> 15;
  int rr  = rem & 32767;
  const float* Wx = (w == 0 ? Wq : Wk) + (size_t)rr * DM;
  const float* q  = qpool + b * DM;
  float s = 0.f;
  for (int i = 0; i < DM; ++i) s += Wx[i] * q[i];
  s *= (1.f / NN);
  int dcol = rr >> 6, e = rr & 63;
  half_t* dst = (w == 0 ? omtQ : omtK);
  dst[((size_t)b * DS + e) * DM + dcol] = (half_t)s;
}

// ---------------- Phi = elu(Qn1 @ Om + bias) + 1  (WMMA, Nc=64) ------------------
__global__ void phi_gemm(const half_t* __restrict__ A, const half_t* __restrict__ OmT,
                         const float* __restrict__ bias, half_t* __restrict__ Phi) {
  int wid  = blockIdx.x * 8 + wave_id();   // 0..511 row tiles
  int row0 = wid * 16;
  int b    = row0 >> 11;
  int lane = lane_id();
  const half_t* W = OmT + (size_t)b * DS * DM;
  v8f acc[4] = {};
  for (int k0 = 0; k0 < DM; k0 += 32) {
    v16h a = load_a(A, DM, row0, k0, lane);
#pragma unroll
    for (int c = 0; c < 4; ++c) acc[c] = wmma16(a, load_b(W, DM, c * 16, k0, lane), acc[c]);
  }
  int colL = lane & 15, hi = lane >> 4;
#pragma unroll
  for (int c = 0; c < 4; ++c)
#pragma unroll
    for (int j = 0; j < 8; ++j) {
      int col = c * 16 + colL;
      float x = acc[c][j] + bias[col];
      float y = (x > 0.f) ? x + 1.f : __expf(x);   // elu(x)+1
      Phi[(size_t)(row0 + j + 8 * hi) * DS + col] = (half_t)y;
    }
}

// ----- Fused W=relu(PhiQ PhiK^T)^2, Norm=rowsum, Attraction=W@Qn1 (flash-style) ---
__global__ void attraction_kernel(const half_t* __restrict__ PhiQ, const half_t* __restrict__ PhiK,
                                  const half_t* __restrict__ Qn1T, float* __restrict__ Attr,
                                  float* __restrict__ Norm) {
  __shared__ __align__(32) half_t sbuf[8][16 * 32];   // per-wave S2 transpose buffer
  int gw    = blockIdx.x * 8 + wave_id();   // 0..2047
  int rt    = gw >> 2;                      // row tile 0..511
  int chunk = gw & 3;                       // 128-wide d chunk
  int b     = (rt * 16) >> 11;
  int n0    = (rt * 16) & (NN - 1);
  int lane  = lane_id();
  int colL  = lane & 15, hi = lane >> 4;
  half_t* lds = sbuf[wave_id()];
  const half_t* phiQ = PhiQ + ((size_t)b * NN + n0) * DS;
  const half_t* phiK = PhiK + (size_t)b * NN * DS;
  const half_t* qT   = Qn1T + (size_t)b * DM * NN + (size_t)(chunk * 128) * NN;

  v16h aq0 = load_a(phiQ, DS, 0, 0, lane);
  v16h aq1 = load_a(phiQ, DS, 0, 32, lane);

  v8f cat[8] = {};
  float part[8] = {0, 0, 0, 0, 0, 0, 0, 0};

  for (int m = 0; m < NN; m += 32) {
    v8f sa = {}, sb = {};
    {
      const half_t* pk0 = phiK + (size_t)m * DS;
      const half_t* pk1 = phiK + (size_t)(m + 16) * DS;
      sa = wmma16(aq0, load_b(pk0, DS, 0, 0, lane), sa);
      sa = wmma16(aq1, load_b(pk0, DS, 0, 32, lane), sa);
      sb = wmma16(aq0, load_b(pk1, DS, 0, 0, lane), sb);
      sb = wmma16(aq1, load_b(pk1, DS, 0, 32, lane), sb);
    }
#pragma unroll
    for (int j = 0; j < 8; ++j) {
      float x = sa[j]; x = x > 0.f ? x : 0.f; x *= x;
      float y = sb[j]; y = y > 0.f ? y : 0.f; y *= y;
      if (chunk == 0) part[j] += x + y;
      lds[(j + 8 * hi) * 32 + colL]      = (half_t)x;
      lds[(j + 8 * hi) * 32 + 16 + colL] = (half_t)y;
    }
    asm volatile("s_wait_dscnt 0" ::: "memory");
    v8h w0 = *(const v8h*)&lds[colL * 32 + hi * 8];
    v8h w1 = *(const v8h*)&lds[colL * 32 + 16 + hi * 8];
    v16h sA;
#pragma unroll
    for (int i = 0; i < 8; ++i) { sA[i] = w0[i]; sA[i + 8] = w1[i]; }
#pragma unroll
    for (int t = 0; t < 8; ++t) {
      const half_t* bp = &qT[(size_t)(t * 16 + colL) * NN + m + hi * 8];
      v8h q0 = *(const v8h*)bp;
      v8h q1 = *(const v8h*)(bp + 16);
      v16h bt;
#pragma unroll
      for (int i = 0; i < 8; ++i) { bt[i] = q0[i]; bt[i + 8] = q1[i]; }
      cat[t] = wmma16(sA, bt, cat[t]);
    }
  }

#pragma unroll
  for (int t = 0; t < 8; ++t)
#pragma unroll
    for (int j = 0; j < 8; ++j)
      Attr[((size_t)b * NN + n0 + j + 8 * hi) * DM + chunk * 128 + t * 16 + colL] = cat[t][j];

  if (chunk == 0) {
#pragma unroll
    for (int j = 0; j < 8; ++j) {
      float s = part[j];
      for (int mk = 8; mk >= 1; mk >>= 1) s += __shfl_xor(s, mk, 32);
      if (colL == 0) Norm[(size_t)b * NN + n0 + j + 8 * hi] = fabsf(s) + 1.f;
    }
  }
}

// ---------------- m = Attr/Norm - Qn1 (f16) --------------------------------------
__global__ void m_kernel(const float* __restrict__ Attr, const float* __restrict__ Norm,
                         const float* __restrict__ qn1f, half_t* __restrict__ mh) {
  size_t i = (size_t)blockIdx.x * 256 + threadIdx.x;
  size_t row = i >> 9;
  mh[i] = (half_t)(Attr[i] / Norm[row] - qn1f[i]);
}

// ------- Q_interact = Q_in + softplus(dt)*(m @ mW^T + mb), async-staged A --------
__global__ void mproj_gemm(const half_t* __restrict__ A, const half_t* __restrict__ W,
                           const float* __restrict__ mb, const float* __restrict__ Qin,
                           const float* __restrict__ dt, float* __restrict__ Qint) {
  __shared__ __align__(16) half_t abuf[2][64 * KCH];
  int lane = lane_id(), tid = threadIdx.x;
  int row0 = blockIdx.x * 64;
  int col0 = (blockIdx.y * 8 + wave_id()) * 32;
  v8f acc[8] = {};
  const int NK = DM / KCH;
  stage_chunk(abuf[0], A, row0, 0, DM, tid);
  for (int kc = 0; kc < NK; ++kc) {
    if (kc + 1 < NK) {
      stage_chunk(abuf[(kc + 1) & 1], A, row0, (kc + 1) * KCH, DM, tid);
      asm volatile("s_wait_asynccnt 0x2" ::: "memory");
    } else {
      asm volatile("s_wait_asynccnt 0x0" ::: "memory");
    }
    __syncthreads();
    const half_t* ab = abuf[kc & 1];
#pragma unroll
    for (int ks = 0; ks < 2; ++ks) {
      int kg = kc * KCH + ks * 32;
      v16h b0 = load_b(W, DM, col0, kg, lane);
      v16h b1 = load_b(W, DM, col0 + 16, kg, lane);
#pragma unroll
      for (int rt = 0; rt < 4; ++rt) {
        v16h a = lds_a(ab, rt, ks * 32, lane);
        acc[rt * 2]     = wmma16(a, b0, acc[rt * 2]);
        acc[rt * 2 + 1] = wmma16(a, b1, acc[rt * 2 + 1]);
      }
    }
    __syncthreads();
  }
  float sp = log1pf(__expf(dt[0]));
  int colL = lane & 15, hi = lane >> 4;
#pragma unroll
  for (int rt = 0; rt < 4; ++rt)
#pragma unroll
    for (int c = 0; c < 2; ++c)
#pragma unroll
      for (int j = 0; j < 8; ++j) {
        int col = col0 + c * 16 + colL;
        size_t o = (size_t)(row0 + rt * 16 + j + 8 * hi) * DM + col;
        Qint[o] = Qin[o] + sp * (acc[rt * 2 + c][j] + mb[col]);
      }
}

// ---------------- LayerNorm 2 -> f16 ---------------------------------------------
__global__ void ln2_kernel(const float* __restrict__ X, const float* __restrict__ g,
                           const float* __restrict__ bta, half_t* __restrict__ out) {
  int row = blockIdx.x;
  const float* x = X + (size_t)row * DM;
  __shared__ float red[8];
  float s = 0.f;
  for (int i = threadIdx.x; i < DM; i += 256) s += x[i];
  for (int m = 16; m >= 1; m >>= 1) s += __shfl_xor(s, m, 32);
  if (lane_id() == 0) red[wave_id()] = s;
  __syncthreads();
  float mu = 0.f;
  for (int i = 0; i < 8; ++i) mu += red[i];
  mu *= (1.f / DM);
  __syncthreads();
  float v = 0.f;
  for (int i = threadIdx.x; i < DM; i += 256) { float d = x[i] - mu; v += d * d; }
  for (int m = 16; m >= 1; m >>= 1) v += __shfl_xor(v, m, 32);
  if (lane_id() == 0) red[wave_id()] = v;
  __syncthreads();
  float var = 0.f;
  for (int i = 0; i < 8; ++i) var += red[i];
  float rstd = rsqrtf(var * (1.f / DM) + 1e-5f);
  for (int i = threadIdx.x; i < DM; i += 256)
    out[(size_t)row * DM + i] = (half_t)((x[i] - mu) * rstd * g[i] + bta[i]);
}

// ---------------- H = silu(G)*U from GU = Qn2 @ W_up^T, async-staged A -----------
__global__ void gu_gemm(const half_t* __restrict__ A, const half_t* __restrict__ W,
                        half_t* __restrict__ H) {
  __shared__ __align__(16) half_t abuf[2][64 * KCH];
  int lane = lane_id(), tid = threadIdx.x;
  int row0 = blockIdx.x * 64;
  int gt   = blockIdx.y * 8 + wave_id();   // 0..127
  int colG = gt * 16, colU = colG + INNER;
  v8f acc[8] = {};
  const int NK = DM / KCH;
  stage_chunk(abuf[0], A, row0, 0, DM, tid);
  for (int kc = 0; kc < NK; ++kc) {
    if (kc + 1 < NK) {
      stage_chunk(abuf[(kc + 1) & 1], A, row0, (kc + 1) * KCH, DM, tid);
      asm volatile("s_wait_asynccnt 0x2" ::: "memory");
    } else {
      asm volatile("s_wait_asynccnt 0x0" ::: "memory");
    }
    __syncthreads();
    const half_t* ab = abuf[kc & 1];
#pragma unroll
    for (int ks = 0; ks < 2; ++ks) {
      int kg = kc * KCH + ks * 32;
      v16h bg = load_b(W, DM, colG, kg, lane);
      v16h bu = load_b(W, DM, colU, kg, lane);
#pragma unroll
      for (int rt = 0; rt < 4; ++rt) {
        v16h a = lds_a(ab, rt, ks * 32, lane);
        acc[rt * 2]     = wmma16(a, bg, acc[rt * 2]);
        acc[rt * 2 + 1] = wmma16(a, bu, acc[rt * 2 + 1]);
      }
    }
    __syncthreads();
  }
  int colL = lane & 15, hi = lane >> 4;
#pragma unroll
  for (int rt = 0; rt < 4; ++rt)
#pragma unroll
    for (int j = 0; j < 8; ++j) {
      float g = acc[rt * 2][j], u = acc[rt * 2 + 1][j];
      float h = g / (1.f + __expf(-g)) * u;
      H[(size_t)(row0 + rt * 16 + j + 8 * hi) * INNER + colG + colL] = (half_t)h;
    }
}

// ---------------- depthwise conv along N, k=3, SAME ------------------------------
__global__ void dwconv_kernel(const half_t* __restrict__ H, const float* __restrict__ k3,
                              half_t* __restrict__ Hc) {
  size_t i = (size_t)blockIdx.x * 256 + threadIdx.x;   // B*N*INNER
  int c = (int)(i & (INNER - 1));
  size_t row = i >> 11;
  int n = (int)(row & (NN - 1));
  float acc = (float)H[i] * k3[c * 3 + 1];
  if (n > 0)      acc += (float)H[i - INNER] * k3[c * 3 + 0];
  if (n < NN - 1) acc += (float)H[i + INNER] * k3[c * 3 + 2];
  Hc[i] = (half_t)acc;
}

// ---------------- out = Q_interact + Hc @ W_down^T, async-staged A ---------------
__global__ void down_gemm(const half_t* __restrict__ A, const half_t* __restrict__ W,
                          const float* __restrict__ Qint, float* __restrict__ out) {
  __shared__ __align__(16) half_t abuf[2][64 * KCH];
  int lane = lane_id(), tid = threadIdx.x;
  int row0 = blockIdx.x * 64;
  int col0 = (blockIdx.y * 8 + wave_id()) * 32;
  v8f acc[8] = {};
  const int NK = INNER / KCH;   // 32
  stage_chunk(abuf[0], A, row0, 0, INNER, tid);
  for (int kc = 0; kc < NK; ++kc) {
    if (kc + 1 < NK) {
      stage_chunk(abuf[(kc + 1) & 1], A, row0, (kc + 1) * KCH, INNER, tid);
      asm volatile("s_wait_asynccnt 0x2" ::: "memory");
    } else {
      asm volatile("s_wait_asynccnt 0x0" ::: "memory");
    }
    __syncthreads();
    const half_t* ab = abuf[kc & 1];
#pragma unroll
    for (int ks = 0; ks < 2; ++ks) {
      int kg = kc * KCH + ks * 32;
      v16h b0 = load_b(W, INNER, col0, kg, lane);
      v16h b1 = load_b(W, INNER, col0 + 16, kg, lane);
#pragma unroll
      for (int rt = 0; rt < 4; ++rt) {
        v16h a = lds_a(ab, rt, ks * 32, lane);
        acc[rt * 2]     = wmma16(a, b0, acc[rt * 2]);
        acc[rt * 2 + 1] = wmma16(a, b1, acc[rt * 2 + 1]);
      }
    }
    __syncthreads();
  }
  int colL = lane & 15, hi = lane >> 4;
#pragma unroll
  for (int rt = 0; rt < 4; ++rt)
#pragma unroll
    for (int c = 0; c < 2; ++c)
#pragma unroll
      for (int j = 0; j < 8; ++j) {
        size_t o = (size_t)(row0 + rt * 16 + j + 8 * hi) * DM + col0 + c * 16 + colL;
        out[o] = Qint[o] + acc[rt * 2 + c][j];
      }
}

// ---------------- f32 -> f16 weight convert --------------------------------------
__global__ void f32_to_f16(const float* __restrict__ src, half_t* __restrict__ dst, int n) {
  int i = blockIdx.x * 256 + threadIdx.x;
  if (i < n) dst[i] = (half_t)src[i];
}

extern "C" void kernel_launch(void* const* d_in, const int* in_sizes, int n_in,
                              void* d_out, int out_size, void* d_ws, size_t ws_size,
                              hipStream_t stream) {
  const float* Q_in    = (const float*)d_in[0];
  const float* n1g     = (const float*)d_in[1];
  const float* n1b     = (const float*)d_in[2];
  const float* Wq      = (const float*)d_in[3];
  const float* Wk      = (const float*)d_in[4];
  const float* B_Q     = (const float*)d_in[5];
  const float* B_K     = (const float*)d_in[6];
  const float* m_W     = (const float*)d_in[7];
  const float* m_b     = (const float*)d_in[8];
  const float* dt      = (const float*)d_in[9];
  const float* n2g     = (const float*)d_in[10];
  const float* n2b     = (const float*)d_in[11];
  const float* W_up    = (const float*)d_in[12];
  const float* dw_k    = (const float*)d_in[13];
  const float* W_down  = (const float*)d_in[14];

  char* ws = (char*)d_ws;
  size_t off = 0;
  auto carve = [&](size_t bytes) -> char* {
    char* p = ws + off;
    off = (off + bytes + 255) & ~(size_t)255;
    return p;
  };
  float*  qn1f  = (float*) carve((size_t)BN * DM * 4);
  half_t* qn1h  = (half_t*)carve((size_t)BN * DM * 2);
  half_t* qn1t  = (half_t*)carve((size_t)BN * DM * 2);
  float*  qpool = (float*) carve((size_t)4 * DM * 4);
  half_t* omtQ  = (half_t*)carve((size_t)4 * DS * DM * 2);
  half_t* omtK  = (half_t*)carve((size_t)4 * DS * DM * 2);
  half_t* phiQ  = (half_t*)carve((size_t)BN * DS * 2);
  half_t* phiK  = (half_t*)carve((size_t)BN * DS * 2);
  float*  attr  = (float*) carve((size_t)BN * DM * 4);
  float*  nrm   = (float*) carve((size_t)BN * 4);
  half_t* mh    = (half_t*)carve((size_t)BN * DM * 2);
  float*  qint  = (float*) carve((size_t)BN * DM * 4);
  half_t* qn2h  = (half_t*)carve((size_t)BN * DM * 2);
  half_t* Hbuf  = (half_t*)carve((size_t)BN * INNER * 2);
  half_t* Hc    = (half_t*)carve((size_t)BN * INNER * 2);
  half_t* mWh   = (half_t*)carve((size_t)DM * DM * 2);
  half_t* Wuph  = (half_t*)carve((size_t)2 * INNER * DM * 2);
  half_t* Wdnh  = (half_t*)carve((size_t)DM * INNER * 2);

  f32_to_f16<<<(DM * DM) / 256, 256, 0, stream>>>(m_W, mWh, DM * DM);
  f32_to_f16<<<(2 * INNER * DM) / 256, 256, 0, stream>>>(W_up, Wuph, 2 * INNER * DM);
  f32_to_f16<<<(DM * INNER) / 256, 256, 0, stream>>>(W_down, Wdnh, DM * INNER);

  hipMemsetAsync(qpool, 0, (size_t)4 * DM * 4, stream);
  ln1_kernel<<<BN, 256, 0, stream>>>(Q_in, n1g, n1b, qn1f, qn1h, qn1t, qpool);

  omega_kernel<<<(2 * 4 * DM * DS) / 256, 256, 0, stream>>>(Wq, Wk, qpool, omtQ, omtK);

  phi_gemm<<<BN / 16 / 8, 256, 0, stream>>>(qn1h, omtQ, B_Q, phiQ);
  phi_gemm<<<BN / 16 / 8, 256, 0, stream>>>(qn1h, omtK, B_K, phiK);

  attraction_kernel<<<(BN / 16) * 4 / 8, 256, 0, stream>>>(phiQ, phiK, qn1t, attr, nrm);

  m_kernel<<<((size_t)BN * DM) / 256, 256, 0, stream>>>(attr, nrm, qn1f, mh);

  mproj_gemm<<<dim3(BN / 64, DM / 32 / 8), 256, 0, stream>>>(mh, mWh, m_b, Q_in, dt, qint);

  ln2_kernel<<<BN, 256, 0, stream>>>(qint, n2g, n2b, qn2h);

  gu_gemm<<<dim3(BN / 64, INNER / 16 / 8), 256, 0, stream>>>(qn2h, Wuph, Hbuf);

  dwconv_kernel<<<((size_t)BN * INNER) / 256, 256, 0, stream>>>(Hbuf, dw_k, Hc);

  down_gemm<<<dim3(BN / 64, DM / 32 / 8), 256, 0, stream>>>(Hc, Wdnh, qint, (float*)d_out);
}
